// EdgeGNN_86723979641091
// MI455X (gfx1250) — matrix operand order repaired and checked by reading, not compiled, not run on hardware
//
#include <hip/hip_runtime.h>
#include <hip/hip_bf16.h>

// ---------------------------------------------------------------------------
// EdgeGNN for MI455X (gfx1250): GEMMs via v_wmma_f32_16x16x32_bf16; weights
// pre-converted to transposed bf16 and staged into LDS per K-chunk with the
// Tensor Data Mover (tensor_load_to_lds + s_wait_tensorcnt). Activations stay
// fp32 in HBM (scatter-add atomics), converted to bf16 during LDS staging.
// Edge MLP fully fused (no 307MB hidden buffer).
// ---------------------------------------------------------------------------

typedef __attribute__((ext_vector_type(16))) __bf16 v16bf;
typedef __attribute__((ext_vector_type(8)))  float  v8f;
typedef unsigned u32x4 __attribute__((ext_vector_type(4)));
typedef unsigned u32x8 __attribute__((ext_vector_type(8)));

#define HIDN  256
#define EMBD  128
#define NCLS  4

// round-to-nearest-even fp32 -> bf16
__device__ __forceinline__ __bf16 f2bf(float f) {
  unsigned u = __builtin_bit_cast(unsigned, f);
  unsigned r = (u + 0x7FFFu + ((u >> 16) & 1u)) >> 16;
  unsigned short s = (unsigned short)r;
  return __builtin_bit_cast(__bf16, s);
}

// A-fragment K index (16x32 bf16 A layout, ISA 7.12.2):
// half=0: i=0..7 -> K0..7, i=8..15 -> K16..23 ; half=1: +8
__device__ __forceinline__ int akmap(int i, int half) {
  return (i & 7) + ((i >> 3) << 4) + (half << 3);
}

// ---------------------------------------------------------------------------
// TDM: DMA a 2-D tile (32 k-elems x 256 cols, bf16) from a transposed weight
// matrix WT[256][K] (row length K elements) into LDS, row-major-compacted so
// LDS layout is [col][32k] — exactly the B-fragment layout.
// D# per CDNA5 ISA (groups 2/3 NULL: 2-D tensor).
// Issue from ONE wave only (TENSORcnt is per-wave); caller barriers after.
// ---------------------------------------------------------------------------
__device__ __forceinline__ void tdm_load_w_chunk(unsigned lds_off,
                                                 const __bf16* gsrc,
                                                 unsigned rowlen) {
  unsigned long long ga = (unsigned long long)(size_t)gsrc;
  u32x4 g0;
  g0[0] = 1u;                                   // count=1, user mode
  g0[1] = lds_off;                              // lds_addr (bytes)
  g0[2] = (unsigned)ga;                         // global_addr[31:0]
  g0[3] = (unsigned)(ga >> 32) | 0x80000000u;   // global_addr[56:32] | type=2
  u32x8 g1;
  g1[0] = 0x00010000u;                          // wg_mask=0, data_size=1 (2B)
  g1[1] = (rowlen & 0xFFFFu) << 16;             // tensor_dim0[15:0]
  g1[2] = ((rowlen >> 16) & 0xFFFFu) | (256u << 16); // dim0 hi | tensor_dim1=256
  g1[3] = (32u << 16);                          // tile_dim0 = 32
  g1[4] = 256u;                                 // tile_dim1 = 256, tile_dim2=0
  g1[5] = rowlen;                               // tensor_dim0_stride[31:0]
  g1[6] = 0u;                                   // stride hi, dim1_stride lo
  g1[7] = 0u;
  asm volatile("tensor_load_to_lds %0, %1" :: "s"(g0), "s"(g1) : "memory");
  __builtin_amdgcn_s_wait_tensorcnt(0);
}

// ---------------------------------------------------------------------------
__global__ void zero_kernel(float* __restrict__ p, long long n) {
  long long i = (long long)blockIdx.x * blockDim.x + threadIdx.x;
  if (i < n) p[i] = 0.f;
}

__global__ void degree_kernel(const int* __restrict__ dst, float* __restrict__ deg, int E) {
  int e = blockIdx.x * blockDim.x + threadIdx.x;
  if (e < E) atomicAdd(&deg[dst[e]], 1.0f);
}

__global__ void invdeg_kernel(float* __restrict__ deg, int N) {
  int i = blockIdx.x * blockDim.x + threadIdx.x;
  if (i < N) deg[i] = 1.0f / fmaxf(deg[i], 1.0f);
}

// convert W[K][256] fp32 -> WT[256][K] bf16 (transposed, coalesced reads)
__global__ void wconv_kernel(const float* __restrict__ W, __bf16* __restrict__ WT, int K) {
  int i = blockIdx.x * blockDim.x + threadIdx.x;
  if (i < K * 256) {
    int k = i >> 8, c = i & 255;
    WT[(size_t)c * K + k] = f2bf(W[i]);
  }
}

// scatter-add of 256-wide features: one thread per (edge, 4 floats)
__global__ void scatter_kernel(const int* __restrict__ src, const int* __restrict__ dst,
                               const float* __restrict__ x, float* __restrict__ agg, int E) {
  long long idx = (long long)blockIdx.x * blockDim.x + threadIdx.x;
  int e = (int)(idx >> 6);
  if (e >= E) return;
  int c = (int)(idx & 63) << 2;
  int s = src[e], d = dst[e];
  const float4 v = *(const float4*)(x + (size_t)s * HIDN + c);
  float* a = agg + (size_t)d * HIDN + c;
  atomicAdd(a + 0, v.x);
  atomicAdd(a + 1, v.y);
  atomicAdd(a + 2, v.z);
  atomicAdd(a + 3, v.w);
}

// ---------------------------------------------------------------------------
// WMMA GEMM: Out[M,256] = A1*scale @ W1T' (+ A2 @ W2T') + bias, optional
// out = relu(.) + A2 residual. Weights are pre-transposed bf16 WT[256][K].
// Block: 256 threads (8 waves) -> 64 rows. Wave (mt,nh): mt=wave>>1 picks
// 16-row tile, nh=wave&1 picks 128-col half.
// ---------------------------------------------------------------------------
template <int K, bool DUAL, bool RELURES>
__global__ void gemm_kernel(const float* __restrict__ A1,
                            const float* __restrict__ scaleA1,   // nullable
                            const __bf16* __restrict__ WT1,      // [256][K]
                            const float* __restrict__ A2,        // DUAL input + residual
                            const __bf16* __restrict__ WT2,      // [256][K]
                            const float* __restrict__ bias,
                            float* __restrict__ Out, int M) {
  __shared__ __bf16 As1[64 * 40];        // 64 rows x 32 k (pad 40)
  __shared__ __bf16 As2[64 * 40];
  __shared__ __bf16 Ws1[256 * 32];       // [col][k] — TDM destination
  __shared__ __bf16 Ws2[256 * 32];

  const int tid  = threadIdx.x;
  const int wave = tid >> 5, lane = tid & 31;
  const int mt   = wave >> 1;            // 0..3
  const int nh   = wave & 1;             // 0..1
  const int half = lane >> 4, l16 = lane & 15;
  const int row0 = blockIdx.x * 64;
  const bool hasScale = (scaleA1 != nullptr);
  const unsigned ws1_off = (unsigned)(size_t)(&Ws1[0]);
  const unsigned ws2_off = (unsigned)(size_t)(&Ws2[0]);

  v8f zero8 = {0.f, 0.f, 0.f, 0.f, 0.f, 0.f, 0.f, 0.f};
  v8f acc[8];
#pragma unroll
  for (int t = 0; t < 8; ++t) acc[t] = zero8;

  for (int kb = 0; kb < K; kb += 32) {
    // --- stage A tiles (fp32 -> bf16), float4 vectorized ---
    for (int i = tid; i < 64 * 8; i += 256) {
      int r = i >> 3, c = (i & 7) << 2;
      int gr = row0 + r;
      float4 f1 = {0.f, 0.f, 0.f, 0.f}, f2 = {0.f, 0.f, 0.f, 0.f};
      if (gr < M) {
        f1 = *(const float4*)(A1 + (size_t)gr * K + kb + c);
        if (hasScale) {
          float s = scaleA1[gr];
          f1.x *= s; f1.y *= s; f1.z *= s; f1.w *= s;
        }
        if (DUAL) f2 = *(const float4*)(A2 + (size_t)gr * K + kb + c);
      }
      __bf16* p1 = &As1[r * 40 + c];
      p1[0] = f2bf(f1.x); p1[1] = f2bf(f1.y); p1[2] = f2bf(f1.z); p1[3] = f2bf(f1.w);
      if (DUAL) {
        __bf16* p2 = &As2[r * 40 + c];
        p2[0] = f2bf(f2.x); p2[1] = f2bf(f2.y); p2[2] = f2bf(f2.z); p2[3] = f2bf(f2.w);
      }
    }
    // --- stage weight chunk via Tensor Data Mover (one wave per matrix) ---
    if (wave == 0) tdm_load_w_chunk(ws1_off, WT1 + kb, (unsigned)K);
    if (DUAL && wave == 1) tdm_load_w_chunk(ws2_off, WT2 + kb, (unsigned)K);
    __syncthreads();

    v16bf a1, a2;
#pragma unroll
    for (int i = 0; i < 16; ++i) {
      int k = akmap(i, half);
      a1[i] = As1[(mt * 16 + l16) * 40 + k];
      if (DUAL) a2[i] = As2[(mt * 16 + l16) * 40 + k];
    }
#pragma unroll
    for (int nt = 0; nt < 8; ++nt) {
      int col = nh * 128 + nt * 16 + l16;
      v16bf b1;
#pragma unroll
      for (int i = 0; i < 16; ++i) b1[i] = Ws1[col * 32 + akmap(i, half)];
      acc[nt] = __builtin_amdgcn_wmma_f32_16x16x32_bf16(
          false, a1, false, b1, (short)0, acc[nt], false, false);
      if (DUAL) {
        v16bf b2v;
#pragma unroll
        for (int i = 0; i < 16; ++i) b2v[i] = Ws2[col * 32 + akmap(i, half)];
        acc[nt] = __builtin_amdgcn_wmma_f32_16x16x32_bf16(
            false, a2, false, b2v, (short)0, acc[nt], false, false);
      }
    }
    __syncthreads();
  }

  // --- epilogue: bias (+relu+residual), C/D layout rows r+8*half ---
#pragma unroll
  for (int nt = 0; nt < 8; ++nt) {
    int n = nh * 128 + nt * 16 + l16;
    float bv = bias[n];
#pragma unroll
    for (int r = 0; r < 8; ++r) {
      int gm = row0 + mt * 16 + r + half * 8;
      if (gm < M) {
        float v = acc[nt][r] + bv;
        if (RELURES) v = fmaxf(v, 0.f) + A2[(size_t)gm * HIDN + n];
        Out[(size_t)gm * HIDN + n] = v;
      }
    }
  }
}

// ---------------------------------------------------------------------------
// Fused edge classifier: gather x[src]||x[dst] -> bf16 LDS, WMMA GEMM with
// W1T (K=512, TDM-staged), ReLU+b1, then 256x4 head reduced via LDS atomics.
// Block: 256 threads, 32 edges. Wave (mt,ng): mt=wave>>2 (16-edge tile),
// ng=wave&3 (64 hidden cols).
// ---------------------------------------------------------------------------
__global__ void edge_mlp_kernel(const int* __restrict__ src, const int* __restrict__ dst,
                                const float* __restrict__ x,
                                const __bf16* __restrict__ W1T,  // [256][512]
                                const float* __restrict__ b1,
                                const float* __restrict__ W2, const float* __restrict__ b2,
                                float* __restrict__ logits, int E) {
  __shared__ __bf16 ef[32 * 520];        // 32 edges x 512 feat (pad 520)
  __shared__ __bf16 Wl[256 * 32];        // [col][k] — TDM destination
  __shared__ float  W2s[256 * 4];
  __shared__ float  lgs[32 * 4];

  const int tid  = threadIdx.x;
  const int wave = tid >> 5, lane = tid & 31;
  const int mt   = wave >> 2;            // 0..1
  const int ng   = wave & 3;             // 0..3
  const int half = lane >> 4, l16 = lane & 15;
  const int e0   = blockIdx.x * 32;
  const unsigned wl_off = (unsigned)(size_t)(&Wl[0]);

  // gather edge features (concat x[src], x[dst]) as bf16, float4 vectorized
  for (int i = tid; i < 32 * 128; i += 256) {
    int e = i >> 7, c = (i & 127) << 2;
    int ge = e0 + e;
    float4 f = {0.f, 0.f, 0.f, 0.f};
    if (ge < E) {
      int node = (c < 256) ? src[ge] : dst[ge];
      f = *(const float4*)(x + (size_t)node * HIDN + (c & 255));
    }
    __bf16* p = &ef[e * 520 + c];
    p[0] = f2bf(f.x); p[1] = f2bf(f.y); p[2] = f2bf(f.z); p[3] = f2bf(f.w);
  }
  for (int i = tid; i < 256 * 4; i += 256) W2s[i] = W2[i];
  if (tid < 32 * 4) lgs[tid] = 0.f;

  v8f zero8 = {0.f, 0.f, 0.f, 0.f, 0.f, 0.f, 0.f, 0.f};
  v8f acc[4];
#pragma unroll
  for (int t = 0; t < 4; ++t) acc[t] = zero8;

  for (int kb = 0; kb < 512; kb += 32) {
    __syncthreads();                     // covers ef/W2s staging + Wl reuse
    if (wave == 0) tdm_load_w_chunk(wl_off, W1T + kb, 512u);
    __syncthreads();

    v16bf a;
#pragma unroll
    for (int i = 0; i < 16; ++i)
      a[i] = ef[(mt * 16 + l16) * 520 + kb + akmap(i, half)];
#pragma unroll
    for (int nt = 0; nt < 4; ++nt) {
      int col = ng * 64 + nt * 16 + l16;
      v16bf b;
#pragma unroll
      for (int i = 0; i < 16; ++i) b[i] = Wl[col * 32 + akmap(i, half)];
      acc[nt] = __builtin_amdgcn_wmma_f32_16x16x32_bf16(
          false, a, false, b, (short)0, acc[nt], false, false);
    }
  }

  // hidden = relu(acc + b1); partial logits = hidden @ W2 (per-lane)
  float plog[8][4];
#pragma unroll
  for (int r = 0; r < 8; ++r)
#pragma unroll
    for (int c = 0; c < 4; ++c) plog[r][c] = 0.f;

#pragma unroll
  for (int nt = 0; nt < 4; ++nt) {
    int hc = ng * 64 + nt * 16 + l16;
    float hb = b1[hc];
    float w0 = W2s[hc * 4 + 0], w1 = W2s[hc * 4 + 1];
    float w2 = W2s[hc * 4 + 2], w3 = W2s[hc * 4 + 3];
#pragma unroll
    for (int r = 0; r < 8; ++r) {
      float hv = fmaxf(acc[nt][r] + hb, 0.f);
      plog[r][0] += hv * w0;
      plog[r][1] += hv * w1;
      plog[r][2] += hv * w2;
      plog[r][3] += hv * w3;
    }
  }
#pragma unroll
  for (int r = 0; r < 8; ++r) {
    int m = mt * 16 + r + half * 8;
#pragma unroll
    for (int c = 0; c < 4; ++c) atomicAdd(&lgs[m * 4 + c], plog[r][c]);
  }
  __syncthreads();
  if (tid < 32 * 4) {
    int e = tid >> 2, c = tid & 3;
    int ge = e0 + e;
    if (ge < E) logits[(size_t)ge * NCLS + c] = lgs[tid] + b2[c];
  }
}

// ---------------------------------------------------------------------------
extern "C" void kernel_launch(void* const* d_in, const int* in_sizes, int n_in,
                              void* d_out, int out_size, void* d_ws, size_t ws_size,
                              hipStream_t stream) {
  const int*   edge_index = (const int*)d_in[0];
  const float* node_emb   = (const float*)d_in[1];
  const float* proj_W     = (const float*)d_in[2];
  const float* proj_b     = (const float*)d_in[3];
  const float* lin_l_W    = (const float*)d_in[4];
  const float* lin_l_b    = (const float*)d_in[5];
  const float* lin_r_W    = (const float*)d_in[6];
  const float* W1         = (const float*)d_in[7];
  const float* b1         = (const float*)d_in[8];
  const float* W2         = (const float*)d_in[9];
  const float* b2         = (const float*)d_in[10];

  const int E = in_sizes[0] / 2;       // 300000
  const int N = in_sizes[1] / EMBD;    // 100000

  const int* src = edge_index;
  const int* dst = edge_index + E;

  float* logits = (float*)d_out;
  float* x0     = (float*)d_out + (size_t)E * NCLS;   // x region of output

  char*  ws     = (char*)d_ws;
  float*  x1     = (float*)ws;   ws += sizeof(float) * (size_t)N * HIDN;
  float*  agg    = (float*)ws;   ws += sizeof(float) * (size_t)N * HIDN;
  float*  invdeg = (float*)ws;   ws += sizeof(float) * (size_t)N;
  __bf16* projWT = (__bf16*)ws;  ws += sizeof(__bf16) * 256 * EMBD;
  __bf16* linLWT = (__bf16*)ws;  ws += sizeof(__bf16) * 3 * 256 * HIDN;
  __bf16* linRWT = (__bf16*)ws;  ws += sizeof(__bf16) * 3 * 256 * HIDN;
  __bf16* W1T    = (__bf16*)ws;  ws += sizeof(__bf16) * 256 * 512;

  // weight pre-transpose + bf16 conversion (tiny)
  wconv_kernel<<<(EMBD * 256 + 255) / 256, 256, 0, stream>>>(proj_W, projWT, EMBD);
  for (int l = 0; l < 3; ++l) {
    wconv_kernel<<<(HIDN * 256 + 255) / 256, 256, 0, stream>>>(
        lin_l_W + (size_t)l * HIDN * HIDN, linLWT + (size_t)l * 256 * HIDN, HIDN);
    wconv_kernel<<<(HIDN * 256 + 255) / 256, 256, 0, stream>>>(
        lin_r_W + (size_t)l * HIDN * HIDN, linRWT + (size_t)l * 256 * HIDN, HIDN);
  }
  wconv_kernel<<<(512 * 256 + 255) / 256, 256, 0, stream>>>(W1, W1T, 512);

  // in-degree -> 1/max(deg,1)
  zero_kernel<<<(N + 255) / 256, 256, 0, stream>>>(invdeg, (long long)N);
  degree_kernel<<<(E + 255) / 256, 256, 0, stream>>>(dst, invdeg, E);
  invdeg_kernel<<<(N + 255) / 256, 256, 0, stream>>>(invdeg, N);

  // x0 = node_emb @ proj_W + proj_b
  const int gb = (N + 63) / 64;
  gemm_kernel<EMBD, false, false><<<gb, 256, 0, stream>>>(
      node_emb, nullptr, projWT, nullptr, nullptr, proj_b, x0, N);

  float* xin = x0;
  float* xout = x1;
  for (int layer = 0; layer < 3; ++layer) {
    long long nh = (long long)N * HIDN;
    zero_kernel<<<(unsigned)((nh + 255) / 256), 256, 0, stream>>>(agg, nh);
    scatter_kernel<<<(unsigned)(((long long)E * 64 + 255) / 256), 256, 0, stream>>>(
        src, dst, xin, agg, E);
    // xout = relu(agg*invdeg @ lin_l + b_l + xin @ lin_r) + xin
    gemm_kernel<HIDN, true, true><<<gb, 256, 0, stream>>>(
        agg, invdeg,
        linLWT + (size_t)layer * 256 * HIDN,
        xin,
        linRWT + (size_t)layer * 256 * HIDN,
        lin_l_b + (size_t)layer * HIDN,
        xout, N);
    float* t = xin; xin = xout; xout = t;
  }
  // final x is in x1 (ws); publish into d_out's x region
  if (xin != x0)
    hipMemcpyAsync(x0, xin, sizeof(float) * (size_t)N * HIDN,
                   hipMemcpyDeviceToDevice, stream);

  // fused edge classifier
  edge_mlp_kernel<<<(E + 31) / 32, 256, 0, stream>>>(
      src, dst, xin, W1T, b1, W2, b2, logits, E);
}
